// PairWeightedAveraging_1898375545300
// MI455X (gfx1250) — compile-verified, matrix operand order checked by reading, not run on hardware
//
#include <hip/hip_runtime.h>
#include <hip/hip_bf16.h>

typedef __attribute__((ext_vector_type(16))) _Float16 v16h;
typedef __attribute__((ext_vector_type(8)))  _Float16 v8h;
typedef __attribute__((ext_vector_type(8)))  float    v8f;

#define DEV static __device__ __forceinline__

// ---------- helpers ----------
DEV v16h cat8(v8h a, v8h b) {
    v16h r;
#pragma unroll
    for (int i = 0; i < 8; ++i) { r[i] = a[i]; r[i + 8] = b[i]; }
    return r;
}

// A/B fragment for V_WMMA_F32_16X16X32_F16, operand row/col contiguous in memory.
// laneSel = (lane>>4)*8 ; per-lane K chunks [k0+laneSel, +8) and [k0+laneSel+16, +8)
DEV v16h load_frag_h(const _Float16* row, int laneSel, int k0) {
    v8h c0 = *(const v8h*)(row + k0 + laneSel);
    v8h c1 = *(const v8h*)(row + k0 + laneSel + 16);
    return cat8(c0, c1);
}

DEV v8f wmma_f16(v16h a, v16h b, v8f c) {
    return __builtin_amdgcn_wmma_f32_16x16x32_f16(false, a, false, b, (short)0, c, false, false);
}

DEV float wave_sum(float x) {
#pragma unroll
    for (int o = 16; o > 0; o >>= 1) x += __shfl_xor(x, o, 32);
    return x;
}
DEV float wave_max(float x) {
#pragma unroll
    for (int o = 16; o > 0; o >>= 1) x = fmaxf(x, __shfl_xor(x, o, 32));
    return x;
}

// ---------- constants ----------
// B=1, S=256, N=512, C_M=64, C_Z=128, C_H=32, H=8, H*C_H=256
#define N_DIM   512
#define S_DIM   256
#define CM      64
#define CZ      128
#define HE      256

// ---------- kernel 0: one-shot weight conversion fp32 -> f16 ----------
// W_m, W_g: (256,64); W_o: (64,256)  -> all 16384 elements, same flat size
__global__ void k_prep_w(const float* __restrict__ Wm, const float* __restrict__ Wg,
                         const float* __restrict__ Wo,
                         _Float16* __restrict__ wm16, _Float16* __restrict__ wg16,
                         _Float16* __restrict__ wo16) {
    int idx = blockIdx.x * 256 + threadIdx.x;    // 64 blocks * 256 = 16384
    wm16[idx] = (_Float16)Wm[idx];
    wg16[idx] = (_Float16)Wg[idx];
    wo16[idx] = (_Float16)Wo[idx];
}

// ---------- kernel 1: layernorm(m) -> f16 ----------
__global__ void k_ln_m(const float* __restrict__ m, const float* __restrict__ w,
                       const float* __restrict__ b, _Float16* __restrict__ mn) {
    int lane = threadIdx.x & 31;
    int wv   = threadIdx.x >> 5;
    long row = (long)blockIdx.x * 8 + wv;        // 131072 rows
    const float* src = m + row * CM;
    float x0 = src[lane * 2], x1 = src[lane * 2 + 1];
    float s1 = wave_sum(x0 + x1);
    float s2 = wave_sum(x0 * x0 + x1 * x1);
    float mu = s1 * (1.0f / CM);
    float var = s2 * (1.0f / CM) - mu * mu;
    float rs = rsqrtf(var + 1e-5f);
    _Float16* dst = mn + row * CM;
    dst[lane * 2]     = (_Float16)((x0 - mu) * rs * w[lane * 2]     + b[lane * 2]);
    dst[lane * 2 + 1] = (_Float16)((x1 - mu) * rs * w[lane * 2 + 1] + b[lane * 2 + 1]);
}

// ---------- kernel 2: v = mn@Wm^T ; g = sigmoid(mn@Wg^T) ----------
// outputs stored transposed: [h][s][e][n] (n contiguous), f16
__global__ void k_vg(const _Float16* __restrict__ mn,
                     const _Float16* __restrict__ wm16, const _Float16* __restrict__ wg16,
                     _Float16* __restrict__ vbuf, _Float16* __restrict__ gbuf) {
    int lane = threadIdx.x & 31;
    int wv   = threadIdx.x >> 5;
    int ln16 = lane & 15;
    int ls   = (lane >> 4) * 8;
    int rowTile = blockIdx.x * 8 + wv;           // 8192 row tiles
    int r0 = rowTile * 16;

    const _Float16* arow = mn + (long)(r0 + ln16) * CM;
    v16h a0 = load_frag_h(arow, ls, 0);
    v16h a1 = load_frag_h(arow, ls, 32);

    int s  = r0 >> 9;                            // row / 512
    int n0 = (r0 & 511) + ls;                    // + 8*(lane>>4)

    // ---- v path ----
    for (int ct = 0; ct < 16; ++ct) {
        int col = ct * 16 + ln16;                // (h,e) flat in [0,256)
        const _Float16* wrow = wm16 + (long)col * CM;
        v8f c = {};
        c = wmma_f16(a0, load_frag_h(wrow, ls, 0),  c);
        c = wmma_f16(a1, load_frag_h(wrow, ls, 32), c);
        int h = col >> 5, e = col & 31;
        _Float16* dst = vbuf + (((long)(h * S_DIM + s) * 32 + e) * N_DIM + n0);
        v8h outv;
#pragma unroll
        for (int r = 0; r < 8; ++r) outv[r] = (_Float16)c[r];
        *(v8h*)dst = outv;
    }
    // ---- gate path (sigmoid) ----
    for (int ct = 0; ct < 16; ++ct) {
        int col = ct * 16 + ln16;
        const _Float16* wrow = wg16 + (long)col * CM;
        v8f c = {};
        c = wmma_f16(a0, load_frag_h(wrow, ls, 0),  c);
        c = wmma_f16(a1, load_frag_h(wrow, ls, 32), c);
        int h = col >> 5, e = col & 31;
        _Float16* dst = gbuf + (((long)(h * S_DIM + s) * 32 + e) * N_DIM + n0);
        v8h outv;
#pragma unroll
        for (int r = 0; r < 8; ++r)
            outv[r] = (_Float16)(1.0f / (1.0f + __expf(-c[r])));
        *(v8h*)dst = outv;
    }
}

// ---------- kernel 3: LN(z) + pair bias + mask + softmax -> w_f16[h][i][j] ----------
__global__ void k_bias_softmax(const float* __restrict__ z, const float* __restrict__ mask,
                               const float* __restrict__ lzw, const float* __restrict__ lzb,
                               const float* __restrict__ Wz, _Float16* __restrict__ wout) {
    __shared__ float logits[8][512];
    int lane = threadIdx.x & 31;
    int wv   = threadIdx.x >> 5;
    int i    = blockIdx.x;                       // 512 blocks

    float lw[4], lb[4], wz[8][4];
#pragma unroll
    for (int k = 0; k < 4; ++k) { lw[k] = lzw[lane * 4 + k]; lb[k] = lzb[lane * 4 + k]; }
#pragma unroll
    for (int h = 0; h < 8; ++h)
#pragma unroll
        for (int k = 0; k < 4; ++k) wz[h][k] = Wz[h * CZ + lane * 4 + k];

    for (int t = 0; t < 64; ++t) {
        int j = wv + t * 8;
        const float4 xf = *(const float4*)(z + ((long)i * N_DIM + j) * CZ + lane * 4);
        float x[4] = {xf.x, xf.y, xf.z, xf.w};
        float s1 = wave_sum(x[0] + x[1] + x[2] + x[3]);
        float s2 = wave_sum(x[0]*x[0] + x[1]*x[1] + x[2]*x[2] + x[3]*x[3]);
        float mu = s1 * (1.0f / CZ);
        float var = s2 * (1.0f / CZ) - mu * mu;
        float rs = rsqrtf(var + 1e-5f);
        float zn[4];
#pragma unroll
        for (int k = 0; k < 4; ++k) zn[k] = (x[k] - mu) * rs * lw[k] + lb[k];
        float acc[8];
#pragma unroll
        for (int h = 0; h < 8; ++h) {
            float a = zn[0]*wz[h][0] + zn[1]*wz[h][1] + zn[2]*wz[h][2] + zn[3]*wz[h][3];
            acc[h] = wave_sum(a);
        }
        if (lane == 0) {
            float pen = (1.0f - mask[(long)i * N_DIM + j]) * (-1000000.0f);
#pragma unroll
            for (int h = 0; h < 8; ++h) logits[h][j] = acc[h] + pen;
        }
    }
    __syncthreads();

    // wave wv owns head wv: softmax over 512
    float vmax = -3.0e38f;
#pragma unroll
    for (int t = 0; t < 16; ++t) vmax = fmaxf(vmax, logits[wv][lane + t * 32]);
    vmax = wave_max(vmax);
    float ex[16], ssum = 0.0f;
#pragma unroll
    for (int t = 0; t < 16; ++t) { ex[t] = __expf(logits[wv][lane + t * 32] - vmax); ssum += ex[t]; }
    ssum = wave_sum(ssum);
    float inv = 1.0f / ssum;
#pragma unroll
    for (int t = 0; t < 16; ++t)
        wout[((long)wv * N_DIM + i) * N_DIM + lane + t * 32] = (_Float16)(ex[t] * inv);
}

// ---------- kernel 4: o = w @ v (per head), gate, scatter to og[(s,i)][h*32+e] ----------
// each wave: 16 rows x 64 cols (4 accumulators) -> A fragment reused 4x per K step
__global__ void k_pwa(const _Float16* __restrict__ wbuf, const _Float16* __restrict__ vbuf,
                      const _Float16* __restrict__ gbuf, _Float16* __restrict__ og) {
    int lane = threadIdx.x & 31;
    int wv   = threadIdx.x >> 5;
    int ln16 = lane & 15;
    int ls   = (lane >> 4) * 8;

    int blk = blockIdx.x;                // 8 heads * 32 itiles * 16 colgroups = 4096
    int h   = blk >> 9;
    int rem = blk & 511;
    int it  = rem >> 4;
    int cg  = rem & 15;
    int ct0 = (cg * 8 + wv) * 4;         // [0,512) step 4 over (s,e) col tiles
    int i0  = it * 16;

    const _Float16* arow = wbuf + ((long)h * N_DIM + (i0 + ln16)) * N_DIM;

    const _Float16* bcol[4];
#pragma unroll
    for (int t = 0; t < 4; ++t) {
        int ct = ct0 + t;
        int s  = ct >> 1;
        int e0 = (ct & 1) * 16;
        bcol[t] = vbuf + (((long)(h * S_DIM + s) * 32) + e0 + ln16) * N_DIM;
    }

    v8f c[4] = {v8f{}, v8f{}, v8f{}, v8f{}};
    for (int kk = 0; kk < 16; ++kk) {
        int k0 = kk * 32;
        __builtin_prefetch(arow + k0 + 128, 0, 1);
        v16h a = load_frag_h(arow, ls, k0);
#pragma unroll
        for (int t = 0; t < 4; ++t) {
            v16h b = load_frag_h(bcol[t], ls, k0);
            c[t] = wmma_f16(a, b, c[t]);
        }
    }

#pragma unroll
    for (int t = 0; t < 4; ++t) {
        int ct = ct0 + t;
        int s  = ct >> 1;
        int e0 = (ct & 1) * 16;
        // gate: g[h][s][e][i], i contiguous -> one v8h per lane
        const _Float16* gp = gbuf + (((long)(h * S_DIM + s) * 32) + e0 + ln16) * N_DIM + i0 + ls;
        v8h g8 = *(const v8h*)gp;
        int colg = h * 32 + e0 + ln16;                // (h,e) flat in [0,256)
        _Float16* op = og + ((long)s * N_DIM + i0 + ls) * HE + colg;
#pragma unroll
        for (int r = 0; r < 8; ++r)
            op[(long)r * HE] = (_Float16)(c[t][r] * (float)g8[r]);
    }
}

// ---------- kernel 5: out = og @ W_o^T (fp32 out) ----------
__global__ void k_out(const _Float16* __restrict__ og, const _Float16* __restrict__ wo16,
                      float* __restrict__ out) {
    int lane = threadIdx.x & 31;
    int wv   = threadIdx.x >> 5;
    int ln16 = lane & 15;
    int ls   = (lane >> 4) * 8;
    int rowTile = blockIdx.x * 8 + wv;   // 8192 row tiles
    int r0 = rowTile * 16;

    const _Float16* arow = og + (long)(r0 + ln16) * HE;
    v16h af[8];
#pragma unroll
    for (int kk = 0; kk < 8; ++kk) af[kk] = load_frag_h(arow, ls, kk * 32);

    for (int ct = 0; ct < 4; ++ct) {
        int col = ct * 16 + ln16;        // [0,64)
        const _Float16* wrow = wo16 + (long)col * HE;
        v8f c = {};
#pragma unroll
        for (int kk = 0; kk < 8; ++kk)
            c = wmma_f16(af[kk], load_frag_h(wrow, ls, kk * 32), c);
        float* op = out + (long)(r0 + ls) * CM + col;
#pragma unroll
        for (int r = 0; r < 8; ++r) op[(long)r * CM] = c[r];
    }
}

// ---------- launch ----------
extern "C" void kernel_launch(void* const* d_in, const int* in_sizes, int n_in,
                              void* d_out, int out_size, void* d_ws, size_t ws_size,
                              hipStream_t stream) {
    (void)in_sizes; (void)n_in; (void)out_size; (void)ws_size;
    const float* m     = (const float*)d_in[0];
    const float* z     = (const float*)d_in[1];
    const float* mask  = (const float*)d_in[2];
    const float* ln_mw = (const float*)d_in[3];
    const float* ln_mb = (const float*)d_in[4];
    const float* ln_zw = (const float*)d_in[5];
    const float* ln_zb = (const float*)d_in[6];
    const float* W_m   = (const float*)d_in[7];
    const float* W_g   = (const float*)d_in[8];
    const float* W_z   = (const float*)d_in[9];
    const float* W_o   = (const float*)d_in[10];
    float* out = (float*)d_out;

    _Float16* mn   = (_Float16*)d_ws;                 //  8,388,608 halves
    _Float16* wbuf = mn   + 8388608;                  //  2,097,152
    _Float16* vbuf = wbuf + 2097152;                  // 33,554,432
    _Float16* gbuf = vbuf + 33554432;                 // 33,554,432
    _Float16* og   = gbuf + 33554432;                 // 33,554,432
    _Float16* wm16 = og   + 33554432;                 //     16,384
    _Float16* wg16 = wm16 + 16384;                    //     16,384
    _Float16* wo16 = wg16 + 16384;                    //     16,384  (~212 MB total)

    k_prep_w      <<<64,    256, 0, stream>>>(W_m, W_g, W_o, wm16, wg16, wo16);
    k_ln_m        <<<16384, 256, 0, stream>>>(m, ln_mw, ln_mb, mn);
    k_vg          <<<1024,  256, 0, stream>>>(mn, wm16, wg16, vbuf, gbuf);
    k_bias_softmax<<<512,   256, 0, stream>>>(z, mask, ln_zw, ln_zb, W_z, wbuf);
    k_pwa         <<<4096,  256, 0, stream>>>(wbuf, vbuf, gbuf, og);
    k_out         <<<1024,  256, 0, stream>>>(og, wo16, out);
}
